// ContextRNN_12111807774829
// MI455X (gfx1250) — matrix operand order repaired and checked by reading, not compiled, not run on hardware
//
#include <hip/hip_runtime.h>
#include <hip/hip_bf16.h>

// ---------------------------------------------------------------------------
// ContextRNN (bidirectional GRU) for MI455X / gfx1250, wave32 + WMMA bf16.
// Pipeline:
//   K0  cvt_transpose : f32 weights -> bf16, N-major (so B-frags are b128 loads)
//   K1  embed         : gather+mask+sum over M=6  -> embedded bf16 [B*T,512]
//   K2  xs_gemm  (x2) : xs = embedded @ k + b_in  -> bf16 [B*T,1536]  (WMMA)
//   K3  gru_scan      : 4 persistent WGs (2 dirs x 2 batch halves of 16 rows),
//                       per step one 16x1536x512 WMMA GEMM + fused gates,
//                       double-buffered LDS h state (1 barrier/step) + prefetch
//   K4  proj     (x2) : [rows,1024] @ Wk + Wb -> f32 (outputs and hidden)
// ---------------------------------------------------------------------------

typedef __bf16 bf16_t;
typedef __attribute__((ext_vector_type(16))) __bf16 v16bf;
typedef __attribute__((ext_vector_type(8)))  __bf16 v8bf;
typedef __attribute__((ext_vector_type(8)))  float  v8f;

static constexpr int Bb = 32, Tt = 512, Mm = 6, Hh = 512, H3 = 1536;
static constexpr int BT = Bb * Tt;

__device__ __forceinline__ v16bf cat8(v8bf lo, v8bf hi) {
  return __builtin_shufflevector(lo, hi, 0,1,2,3,4,5,6,7,8,9,10,11,12,13,14,15);
}

// A-matrix 16x32 bf16 fragment (ISA 7.12.2): lane = {M=lane&15, half=lane>>4},
// VGPR v<4: K = half*8 + 2v, 2v+1 ; v>=4: K = 16 + half*8 + ...
__device__ __forceinline__ v16bf load_a_frag(const bf16_t* A, int lda, int kb, int lane) {
  const int r = lane & 15, half = lane >> 4;
  const bf16_t* p = A + (size_t)r * lda + kb + half * 8;
  v8bf lo = *(const v8bf*)(p);        // K = kb+half*8 .. +7
  v8bf hi = *(const v8bf*)(p + 16);   // K = kb+16+half*8 .. +7
  return cat8(lo, hi);
}

// B-matrix 32x16 bf16 fragment from N-major storage Bt[N][K]:
// lane = {N=lane&15, half=lane>>4}, VGPR v<4: K = half*8 + 2v, 2v+1 ; v>=4: K = 16 + half*8 + ... (same as A)
__device__ __forceinline__ v16bf load_b_frag(const bf16_t* Bt, int ldk, int kb, int lane) {
  const int c = lane & 15, half = lane >> 4;
  const bf16_t* p = Bt + (size_t)c * ldk + kb + half * 8;
  v8bf lo = *(const v8bf*)(p);
  v8bf hi = *(const v8bf*)(p + 16);
  return cat8(lo, hi);
}

__device__ __forceinline__ v8f wmma_bf16(v16bf a, v16bf b, v8f c) {
  return __builtin_amdgcn_wmma_f32_16x16x32_bf16(false, a, false, b, (short)0, c,
                                                 false, false);
}

// ---------------------------------------------------------------- K0: convert
__global__ void cvt_transpose_kernel(const float* __restrict__ in,
                                     bf16_t* __restrict__ out, int R, int C) {
  int idx = blockIdx.x * blockDim.x + threadIdx.x;
  if (idx >= R * C) return;
  int r = idx / C, c = idx % C;                 // in is [R][C] row-major
  out[(size_t)c * R + r] = (bf16_t)in[idx];     // out is [C][R] (N-major)
}

// ---------------------------------------------------------------- K1: embed
__global__ void embed_kernel(const int* __restrict__ seqs,
                             const float* __restrict__ emb,
                             bf16_t* __restrict__ outbf) {
  const int row = blockIdx.x;            // b*T + t
  const int j = threadIdx.x;             // 256 threads, 2 elems each
  const int* s = seqs + (size_t)row * Mm;
  float a0 = 0.f, a1 = 0.f;
  #pragma unroll
  for (int m = 0; m < Mm; ++m) {
    const int tok = s[m];
    if (tok != 1) {                      // PAD_TOKEN == 1
      const float* e = emb + (size_t)tok * Hh;
      a0 += e[j];
      a1 += e[j + 256];
    }
  }
  outbf[(size_t)row * Hh + j]       = (bf16_t)a0;
  outbf[(size_t)row * Hh + j + 256] = (bf16_t)a1;
}

// ---------------------------------------------------------------- K2: xs GEMM
// C[BT,1536] = A[BT,512] @ k[512,1536] + b_in ;  kT given N-major [1536][512].
// Block: 256 thr = 8 waves (4 M x 2 N), tile 128x64; wave tile 32x32.
__global__ void xs_gemm_kernel(const bf16_t* __restrict__ A,
                               const bf16_t* __restrict__ kT,
                               const float* __restrict__ bin,
                               bf16_t* __restrict__ xs) {
  const int w = threadIdx.x >> 5, lane = threadIdx.x & 31;
  const int wm = w >> 1, wn = w & 1;
  const int rowBase = blockIdx.y * 128 + wm * 32;
  const int colBase = blockIdx.x * 64 + wn * 32;
  const int half = lane >> 4, nl = lane & 15;

  v8f acc[2][2] = {};
  #pragma unroll 4
  for (int ks = 0; ks < 16; ++ks) {
    const int kb = ks * 32;
    v16bf a0 = load_a_frag(A + (size_t)rowBase * Hh,        Hh, kb, lane);
    v16bf a1 = load_a_frag(A + (size_t)(rowBase + 16) * Hh, Hh, kb, lane);
    v16bf b0 = load_b_frag(kT + (size_t)colBase * Hh,        Hh, kb, lane);
    v16bf b1 = load_b_frag(kT + (size_t)(colBase + 16) * Hh, Hh, kb, lane);
    acc[0][0] = wmma_bf16(a0, b0, acc[0][0]);
    acc[0][1] = wmma_bf16(a0, b1, acc[0][1]);
    acc[1][0] = wmma_bf16(a1, b0, acc[1][0]);
    acc[1][1] = wmma_bf16(a1, b1, acc[1][1]);
  }
  #pragma unroll
  for (int mt = 0; mt < 2; ++mt)
    #pragma unroll
    for (int nt = 0; nt < 2; ++nt)
      #pragma unroll
      for (int v = 0; v < 8; ++v) {
        const int m = rowBase + mt * 16 + v + 8 * half;
        const int n = colBase + nt * 16 + nl;
        xs[(size_t)m * H3 + n] = (bf16_t)(acc[mt][nt][v] + bin[n]);
      }
}

// ---------------------------------------------------------------- K3: GRU scan
// 4 blocks: blockIdx.x = {dir,bhalf}. 1024 thr = 32 waves. Wave w owns hidden
// cols [16w,16w+16) and computes its own z/r/h gate tiles; batch half = 16 rows
// = one WMMA M tile. h: f32 in regs (accumulator layout) + bf16 copy in a
// double-buffered LDS tile (A fragment for every wave next step, one barrier
// per step). Keras GRUCell semantics, reset_after=True.
__global__ __launch_bounds__(1024) void gru_scan_kernel(
    const bf16_t* __restrict__ xs_f, const bf16_t* __restrict__ xs_b,
    const bf16_t* __restrict__ rkT_f, const bf16_t* __restrict__ rkT_b,
    const float* __restrict__ bias_f, const float* __restrict__ bias_b,
    const int* __restrict__ lens,
    bf16_t* __restrict__ out_f, bf16_t* __restrict__ out_b,
    bf16_t* __restrict__ hT_f, bf16_t* __restrict__ hT_b) {
  const int dir   = blockIdx.x >> 1;
  const int bhalf = blockIdx.x & 1;
  const bf16_t* xs  = dir ? xs_b  : xs_f;
  const bf16_t* rkT = dir ? rkT_b : rkT_f;
  const float* brec = (dir ? bias_b : bias_f) + H3;   // bias row 1 (recurrent)
  bf16_t* out = dir ? out_b : out_f;
  bf16_t* hT  = dir ? hT_b  : hT_f;

  const int tid = threadIdx.x;
  const int w = tid >> 5, lane = tid & 31;
  const int half = lane >> 4, nl = lane & 15;
  const int j = w * 16 + nl;                 // hidden column owned by this lane
  const int bBase = bhalf * 16;              // global batch offset of this WG

  __shared__ bf16_t hsh[2][16][Hh];          // double-buffered bf16 state

  for (int i = tid; i < 16 * Hh; i += 1024) (&hsh[0][0][0])[i] = (bf16_t)0.f;

  const bf16_t* rkz = rkT + (size_t)(0 * Hh + w * 16) * Hh;
  const bf16_t* rkr = rkT + (size_t)(1 * Hh + w * 16) * Hh;
  const bf16_t* rkh = rkT + (size_t)(2 * Hh + w * 16) * Hh;

  float hreg[8];
  int   lenr[8];
  #pragma unroll
  for (int v = 0; v < 8; ++v) {
    hreg[v] = 0.f;
    lenr[v] = lens[bBase + v + 8 * half];
  }
  const float brz = brec[j], brr = brec[Hh + j], brh = brec[2 * Hh + j];

  __syncthreads();

  for (int step = 0; step < Tt; ++step) {
    const int t = dir ? (Tt - 1 - step) : step;
    const int tn = dir ? (t > 0 ? t - 1 : 0) : (t < Tt - 1 ? t + 1 : t);
    const bf16_t* hcur = &hsh[step & 1][0][0];
    bf16_t* hnext = &hsh[(step + 1) & 1][0][0];

    // prefetch next step's xs gate rows while the GEMM runs (global_prefetch)
    if (nl == 0) {
      #pragma unroll
      for (int v = 0; v < 8; ++v) {
        const size_t rn = ((size_t)(bBase + v + 8 * half) * Tt + tn) * H3 + j;
        __builtin_prefetch(xs + rn, 0, 1);
        __builtin_prefetch(xs + rn + Hh, 0, 1);
        __builtin_prefetch(xs + rn + 2 * Hh, 0, 1);
      }
    }

    v8f az = {}, ar = {}, ah = {};           // inner = h @ rk (bias added later)
    #pragma unroll 4
    for (int ks = 0; ks < 16; ++ks) {
      const int kb = ks * 32;
      v16bf af = load_a_frag(hcur, Hh, kb, lane);
      v16bf bz = load_b_frag(rkz, Hh, kb, lane);
      v16bf br = load_b_frag(rkr, Hh, kb, lane);
      v16bf bh = load_b_frag(rkh, Hh, kb, lane);
      az = wmma_bf16(af, bz, az);
      ar = wmma_bf16(af, br, ar);
      ah = wmma_bf16(af, bh, ah);
    }

    #pragma unroll
    for (int v = 0; v < 8; ++v) {
      const int bl = v + 8 * half;           // local batch row (C-layout M)
      const int bg = bBase + bl;
      const size_t row = (size_t)bg * Tt + t;
      const float xz = (float)xs[row * H3 + j];
      const float xr = (float)xs[row * H3 + Hh + j];
      const float xh = (float)xs[row * H3 + 2 * Hh + j];
      const float z = 1.f / (1.f + __expf(-(xz + az[v] + brz)));
      const float r = 1.f / (1.f + __expf(-(xr + ar[v] + brr)));
      const float hh = tanhf(xh + r * (ah[v] + brh));
      float hn = z * hreg[v] + (1.f - z) * hh;
      if (t >= lenr[v]) hn = hreg[v];        // masked step carries state
      hreg[v] = hn;
      const bf16_t hb = (bf16_t)hn;
      hnext[(size_t)bl * Hh + j] = hb;       // write to the *other* buffer
      out[row * Hh + j] = hb;
    }
    __syncthreads();                         // next-step A fragments are ready
  }

  #pragma unroll
  for (int v = 0; v < 8; ++v) {
    const int bg = bBase + v + 8 * half;
    hT[(size_t)bg * Hh + j] = (bf16_t)hreg[v];
  }
}

// ---------------------------------------------------------------- K4: project
// out[rows,512] = concat(A0,A1)[rows,1024] @ Wk + Wb, f32 result.
// Same tiling as K2; K loop split: first 512 from A0, next 512 from A1.
__global__ void proj_kernel(const bf16_t* __restrict__ A0,
                            const bf16_t* __restrict__ A1,
                            const bf16_t* __restrict__ WkT,   // [512][1024]
                            const float* __restrict__ Wb,
                            float* __restrict__ out, int rows) {
  const int w = threadIdx.x >> 5, lane = threadIdx.x & 31;
  const int wm = w >> 1, wn = w & 1;
  const int rowBase = blockIdx.y * 128 + wm * 32;
  const int colBase = blockIdx.x * 64 + wn * 32;
  if (rowBase >= rows) return;               // hidden pass: only wave row 0..31
  const int half = lane >> 4, nl = lane & 15;

  v8f acc[2][2] = {};
  #pragma unroll 4
  for (int ks = 0; ks < 32; ++ks) {
    const bf16_t* Asrc = (ks < 16) ? A0 : A1;
    const int kb  = ((ks < 16) ? ks : ks - 16) * 32;
    const int kbW = ks * 32;
    v16bf a0 = load_a_frag(Asrc + (size_t)rowBase * Hh,        Hh, kb, lane);
    v16bf a1 = load_a_frag(Asrc + (size_t)(rowBase + 16) * Hh, Hh, kb, lane);
    v16bf b0 = load_b_frag(WkT + (size_t)colBase * 1024,        1024, kbW, lane);
    v16bf b1 = load_b_frag(WkT + (size_t)(colBase + 16) * 1024, 1024, kbW, lane);
    acc[0][0] = wmma_bf16(a0, b0, acc[0][0]);
    acc[0][1] = wmma_bf16(a0, b1, acc[0][1]);
    acc[1][0] = wmma_bf16(a1, b0, acc[1][0]);
    acc[1][1] = wmma_bf16(a1, b1, acc[1][1]);
  }
  #pragma unroll
  for (int mt = 0; mt < 2; ++mt)
    #pragma unroll
    for (int nt = 0; nt < 2; ++nt)
      #pragma unroll
      for (int v = 0; v < 8; ++v) {
        const int m = rowBase + mt * 16 + v + 8 * half;
        const int n = colBase + nt * 16 + nl;
        if (m < rows) out[(size_t)m * Hh + n] = acc[mt][nt][v] + Wb[n];
      }
}

// ---------------------------------------------------------------------------
extern "C" void kernel_launch(void* const* d_in, const int* in_sizes, int n_in,
                              void* d_out, int out_size, void* d_ws, size_t ws_size,
                              hipStream_t stream) {
  const int*   seqs = (const int*)d_in[0];
  const int*   lens = (const int*)d_in[1];
  const float* emb  = (const float*)d_in[2];
  const float* k_f  = (const float*)d_in[3];
  const float* rk_f = (const float*)d_in[4];
  const float* b_f  = (const float*)d_in[5];
  const float* k_b  = (const float*)d_in[6];
  const float* rk_b = (const float*)d_in[7];
  const float* b_b  = (const float*)d_in[8];
  const float* Wk   = (const float*)d_in[9];
  const float* Wb   = (const float*)d_in[10];
  float* out = (float*)d_out;

  // workspace carve-up (bf16 buffers)
  char* p = (char*)d_ws;
  auto take = [&](size_t bytes) { char* q = p; p += (bytes + 255) & ~size_t(255); return q; };
  bf16_t* kT_f   = (bf16_t*)take((size_t)H3 * Hh * 2);
  bf16_t* kT_b   = (bf16_t*)take((size_t)H3 * Hh * 2);
  bf16_t* rkT_f  = (bf16_t*)take((size_t)H3 * Hh * 2);
  bf16_t* rkT_b  = (bf16_t*)take((size_t)H3 * Hh * 2);
  bf16_t* WkT    = (bf16_t*)take((size_t)Hh * 2 * Hh * 2);
  bf16_t* embbf  = (bf16_t*)take((size_t)BT * Hh * 2);
  bf16_t* xs_f   = (bf16_t*)take((size_t)BT * H3 * 2);
  bf16_t* xs_b   = (bf16_t*)take((size_t)BT * H3 * 2);
  bf16_t* out_f  = (bf16_t*)take((size_t)BT * Hh * 2);
  bf16_t* out_b  = (bf16_t*)take((size_t)BT * Hh * 2);
  bf16_t* hT_f   = (bf16_t*)take((size_t)Bb * Hh * 2);
  bf16_t* hT_b   = (bf16_t*)take((size_t)Bb * Hh * 2);

  // K0: weight convert + transpose to N-major bf16
  {
    int n1 = Hh * H3;                       // 512x1536 matrices
    dim3 g((n1 + 255) / 256);
    cvt_transpose_kernel<<<g, 256, 0, stream>>>(k_f,  kT_f,  Hh, H3);
    cvt_transpose_kernel<<<g, 256, 0, stream>>>(rk_f, rkT_f, Hh, H3);
    cvt_transpose_kernel<<<g, 256, 0, stream>>>(k_b,  kT_b,  Hh, H3);
    cvt_transpose_kernel<<<g, 256, 0, stream>>>(rk_b, rkT_b, Hh, H3);
    int n2 = 2 * Hh * Hh;                   // Wk 1024x512
    cvt_transpose_kernel<<<dim3((n2 + 255) / 256), 256, 0, stream>>>(Wk, WkT, 2 * Hh, Hh);
  }

  // K1: embedding gather + pad-mask + sum over M
  embed_kernel<<<BT, 256, 0, stream>>>(seqs, emb, embbf);

  // K2: input GEMMs (both directions)
  {
    dim3 g(H3 / 64, BT / 128);
    xs_gemm_kernel<<<g, 256, 0, stream>>>(embbf, kT_f, b_f, xs_f);
    xs_gemm_kernel<<<g, 256, 0, stream>>>(embbf, kT_b, b_b, xs_b);
  }

  // K3: bidirectional GRU scan — 4 independent persistent workgroups
  gru_scan_kernel<<<4, 1024, 0, stream>>>(xs_f, xs_b, rkT_f, rkT_b,
                                          b_f, b_b, lens,
                                          out_f, out_b, hT_f, hT_b);

  // K4: output projection (outputs then hidden)
  proj_kernel<<<dim3(Hh / 64, BT / 128), 256, 0, stream>>>(
      out_f, out_b, WkT, Wb, out, BT);
  proj_kernel<<<dim3(Hh / 64, 1), 256, 0, stream>>>(
      hT_f, hT_b, WkT, Wb, out + (size_t)BT * Hh, Bb);
}
